// Mymodel_40570261078455
// MI455X (gfx1250) — matrix operand order repaired and checked by reading, not compiled, not run on hardware
//
#include <hip/hip_runtime.h>
#include <cstdint>

// ---------------------------------------------------------------------------
// MI455X (gfx1250) implementation of the pointer-generator seq2seq forward.
// All GEMM-shaped math runs through v_wmma_f32_16x16x32_bf16 (wave32 WMMA),
// persistent-workgroup LSTM kernels keep h in LDS (bf16) and c in registers,
// gates live in LDS (128KB -- CDNA5 WGP has 320KB).
// ---------------------------------------------------------------------------

typedef __attribute__((ext_vector_type(16))) __bf16 v16bf;
typedef __attribute__((ext_vector_type(8)))  __bf16 v8bf;
typedef __attribute__((ext_vector_type(8)))  float  v8f;
typedef __attribute__((ext_vector_type(4)))  float  v4f;

#define DEVINL __device__ __forceinline__

DEVINL v8f wmma_bf16(v16bf a, v16bf b, v8f c) {
  return __builtin_amdgcn_wmma_f32_16x16x32_bf16(false, a, false, b,
                                                 (short)0, c, false, false);
}

DEVINL float sigf(float x) { return 1.0f / (1.0f + expf(-x)); }

// ---- WMMA tile loaders (per CDNA5 ISA 7.12.2 VGPR layouts) ----------------
// A 16x32 bf16 tile, converted on the fly from row-major f32 (lda = K).
// lane: hf = lane>>4, m = lane&15. elems 0..7 -> K = hf*8 + i,
// elems 8..15 -> K = 16 + hf*8 + (i-8).
DEVINL v16bf load_a_f32(const float* A, int lda, int m0, int k0) {
  int lane = threadIdx.x & 31;
  int hf = lane >> 4, m = lane & 15;
  const float* p = A + (size_t)(m0 + m) * lda + k0 + hf * 8;
  v4f u0 = *(const v4f*)(p);
  v4f u1 = *(const v4f*)(p + 4);
  v4f u2 = *(const v4f*)(p + 16);
  v4f u3 = *(const v4f*)(p + 20);
  v16bf r;
#pragma unroll
  for (int i = 0; i < 4; ++i) {
    r[i]      = (__bf16)u0[i];
    r[i + 4]  = (__bf16)u1[i];
    r[i + 8]  = (__bf16)u2[i];
    r[i + 12] = (__bf16)u3[i];
  }
  return r;
}

// A 16x32 bf16 tile from LDS-resident hidden state h[16][512].
DEVINL v16bf load_a_lds(const __bf16* h, int k0) {
  int lane = threadIdx.x & 31;
  int hf = lane >> 4, m = lane & 15;
  const __bf16* p = h + m * 512 + k0 + hf * 8;
  v8bf lo = *(const v8bf*)(p);
  v8bf hi = *(const v8bf*)(p + 16);
  v16bf r;
#pragma unroll
  for (int i = 0; i < 8; ++i) { r[i] = lo[i]; r[i + 8] = hi[i]; }
  return r;
}

// B 32x16 bf16 tile where B = W^T, W row-major [N,K] (ldb = K).
// lane: hf = lane>>4, n = lane&15. elems 0..7 -> K = hf*8 + i, elems 8..15 -> K = 16 + hf*8 + (i-8).
DEVINL v16bf load_b_w(const __bf16* W, int ldb, int n0, int k0) {
  int lane = threadIdx.x & 31;
  int hf = lane >> 4, n = lane & 15;
  const __bf16* p = W + (size_t)(n0 + n) * ldb + k0 + hf * 8;
  v8bf lo = *(const v8bf*)(p);
  v8bf hi = *(const v8bf*)(p + 16);
  v16bf r;
#pragma unroll
  for (int i = 0; i < 8; ++i) { r[i] = lo[i]; r[i + 8] = hi[i]; }
  return r;
}

// C/D 16x16 f32: VGPR r holds row hf*8 + r, col lane&15.
DEVINL void store_c_global(float* C, int ldc, int m0, int n0, v8f acc,
                           const float* bias, bool accum) {
  int lane = threadIdx.x & 31;
  int hf = lane >> 4, n = lane & 15;
  float bv = bias ? bias[n0 + n] : 0.0f;
#pragma unroll
  for (int r = 0; r < 8; ++r) {
    size_t idx = (size_t)(m0 + hf * 8 + r) * ldc + n0 + n;
    float v = acc[r] + bv;
    if (accum) v += C[idx];
    C[idx] = v;
  }
}

// ---------------------------------------------------------------------------
// Generic GEMM: C[M,N] (=|+=) A_f32[M,K] @ W_bf16[N,K]^T (+ bias[N]).
// Block = 8 waves; each wave owns one 16x16 output tile of a 16x128 block.
// ---------------------------------------------------------------------------
__global__ __launch_bounds__(256) void gemm_aw(
    const float* __restrict__ A, const __bf16* __restrict__ W,
    const float* __restrict__ bias, float* __restrict__ C,
    int M, int N, int K, int accum) {
  int wid = threadIdx.x >> 5;
  int m0 = blockIdx.x * 16;
  int n0 = blockIdx.y * 128 + wid * 16;
  if (m0 >= M || n0 >= N) return;
  v8f acc = {};
  for (int k0 = 0; k0 < K; k0 += 32)
    acc = wmma_bf16(load_a_f32(A, K, m0, k0), load_b_w(W, K, n0, k0), acc);
  store_c_global(C, N, m0, n0, acc, bias, accum != 0);
}

// ---------------------------------------------------------------------------
// Elementwise helpers
// ---------------------------------------------------------------------------
__global__ void cvt_f32_bf16_k(const float* __restrict__ s, __bf16* __restrict__ d, int n) {
  int i = blockIdx.x * 256 + threadIdx.x;
  if (i < n) d[i] = (__bf16)s[i];
}

__global__ void tanh_k(const float* __restrict__ s, float* __restrict__ d, int n) {
  int i = blockIdx.x * 256 + threadIdx.x;
  if (i < n) d[i] = tanhf(s[i]);
}

// h_k_attr[a,b,:] = sum_{l<4} h_k[a*4+l, b, :]
__global__ void hk_attr_k(const float* __restrict__ hk, float* __restrict__ out) {
  int i = blockIdx.x * 256 + threadIdx.x;
  if (i >= 16 * 16 * 1024) return;
  int col = i & 1023;
  int b = (i >> 10) & 15;
  int a = i >> 14;
  float s = 0.0f;
#pragma unroll
  for (int l = 0; l < 4; ++l)
    s += hk[(size_t)((a * 4 + l) * 16 + b) * 1024 + col];
  out[i] = s;
}

// ---------------------------------------------------------------------------
// Persistent encoder LSTM: one workgroup per (encoder, direction).
// gates[16,2048] = x_t @ Wih^T + h @ Whh^T computed by 8 waves of WMMA.
// h kept bf16 in LDS; c in registers (32 floats/thread).
// ---------------------------------------------------------------------------
struct EncArgs {
  const float*  x[6];
  const __bf16* Wih[6];
  const __bf16* Whh[6];
  const float*  bias[6];
  float*        hcat[6];  // [L,16,1024], dir selects half
  float*        cell[6];  // [16,1024], dir selects half
  int           L[6];
  int           dir[6];
};

__global__ __launch_bounds__(256) void lstm_enc(EncArgs a) {
  extern __shared__ char smem[];
  __bf16* hS   = (__bf16*)smem;                 // 16*512 bf16 (16KB)
  float*  gates = (float*)(smem + 16 * 512 * 2); // 16*2048 f32 (128KB)

  int g = blockIdx.x;
  const float*  x    = a.x[g];
  const __bf16* Wih  = a.Wih[g];
  const __bf16* Whh  = a.Whh[g];
  const float*  bias = a.bias[g];
  float* hcat = a.hcat[g];
  float* cell = a.cell[g];
  int L = a.L[g], dir = a.dir[g];

  int tid = threadIdx.x, wid = tid >> 5, lane = tid & 31;
  for (int i = tid; i < 16 * 512; i += 256) hS[i] = (__bf16)0.0f;
  float creg[32];
#pragma unroll
  for (int u = 0; u < 32; ++u) creg[u] = 0.0f;
  __syncthreads();

  for (int step = 0; step < L; ++step) {
    int t = dir ? (L - 1 - step) : step;
    const float* xt = x + (size_t)t * 16 * 256;

#pragma unroll 1
    for (int nt = 0; nt < 16; ++nt) {
      int n0 = wid * 256 + nt * 16;
      v8f acc = {};
#pragma unroll
      for (int k0 = 0; k0 < 256; k0 += 32)
        acc = wmma_bf16(load_a_f32(xt, 256, 0, k0), load_b_w(Wih, 256, n0, k0), acc);
#pragma unroll
      for (int k0 = 0; k0 < 512; k0 += 32)
        acc = wmma_bf16(load_a_lds(hS, k0), load_b_w(Whh, 512, n0, k0), acc);
      int hf = lane >> 4, n = lane & 15;
#pragma unroll
      for (int r = 0; r < 8; ++r)
        gates[(hf * 8 + r) * 2048 + n0 + n] = acc[r];
    }
    __syncthreads();

#pragma unroll 1
    for (int u = 0; u < 32; ++u) {
      int idx = tid * 32 + u;
      int b = idx >> 9, j = idx & 511;
      const float* gb = gates + b * 2048;
      float gi = gb[j]        + bias[j];
      float gf = gb[512 + j]  + bias[512 + j];
      float gg = gb[1024 + j] + bias[1024 + j];
      float go = gb[1536 + j] + bias[1536 + j];
      float c = sigf(gf) * creg[u] + sigf(gi) * tanhf(gg);
      creg[u] = c;
      float h = sigf(go) * tanhf(c);
      hcat[((size_t)t * 16 + b) * 1024 + dir * 512 + j] = h;
      hS[b * 512 + j] = (__bf16)h;
    }
    __syncthreads();
  }
#pragma unroll 1
  for (int u = 0; u < 32; ++u) {
    int idx = tid * 32 + u;
    int b = idx >> 9, j = idx & 511;
    cell[b * 1024 + dir * 512 + j] = creg[u];
  }
}

// ---------------------------------------------------------------------------
// Persistent attention decoder: one workgroup per decoder (s, k, v).
// Per step: hv = h@v_a^T (WMMA) -> scores tanh-dot u_a -> softmax over L ->
// context ct -> gates = t_emb@Wih^T + h@Whh^T (WMMA) -> LSTMCell update.
// Attention uses h BEFORE the cell update, as in the reference.
// ---------------------------------------------------------------------------
struct DecArgs {
  const float* h0[3];
  const float* c0[3];
  const float* waH[3];   // [L*16, 512]
  const float* inH[3];   // [L*16, 1024]
  float*       alpha[3]; // [T, L, 16]
  float*       cOut[3];  // [T,16,1024]
  float*       sOut[3];  // [T,16,512]
  int          L[3];
  const __bf16* v_a;     // [512,512]
  const float*  u_a;     // [512]
  const __bf16* dWih;    // [2048,256]
  const __bf16* dWhh;    // [2048,512]
  const float*  db;      // [2048]
  const float*  temb;    // [T,16,256]
};

__global__ __launch_bounds__(256) void lstm_dec(DecArgs a) {
  extern __shared__ char smem[];
  __bf16* hS    = (__bf16*)smem;                          // 16KB
  float*  gates = (float*)(smem + 16384);                 // 128KB
  float*  hv    = gates;                                  // alias (16*512 f32)
  float*  scores= (float*)(smem + 16384 + 131072);        // 96*16
  float*  alphL = scores + 96 * 16;                       // 96*16

  int g = blockIdx.x;
  int L = a.L[g];
  const float* waH = a.waH[g];
  const float* inH = a.inH[g];
  int tid = threadIdx.x, wid = tid >> 5, lane = tid & 31;

  for (int i = tid; i < 16 * 512; i += 256) hS[i] = (__bf16)a.h0[g][i];
  float creg[32];
#pragma unroll
  for (int u = 0; u < 32; ++u) creg[u] = a.c0[g][tid * 32 + u];
  __syncthreads();

  for (int t = 0; t < 32; ++t) {
    // (a) hv = h @ v_a^T   [16,512]
#pragma unroll 1
    for (int nt = 0; nt < 4; ++nt) {
      int n0 = wid * 64 + nt * 16;
      v8f acc = {};
#pragma unroll
      for (int k0 = 0; k0 < 512; k0 += 32)
        acc = wmma_bf16(load_a_lds(hS, k0), load_b_w(a.v_a, 512, n0, k0), acc);
      int hf = lane >> 4, n = lane & 15;
#pragma unroll
      for (int r = 0; r < 8; ++r)
        hv[(hf * 8 + r) * 512 + n0 + n] = acc[r];
    }
    __syncthreads();

    // (b) scores[l,b] = tanh(waH[l,b,:] + hv[b,:]) . u_a
    for (int row = wid; row < L * 16; row += 8) {
      int b = row & 15;
      const float* wrow = waH + (size_t)row * 512;
      const float* hrow = hv + b * 512;
      float s = 0.0f;
      for (int hh = lane; hh < 512; hh += 32)
        s += tanhf(wrow[hh] + hrow[hh]) * a.u_a[hh];
#pragma unroll
      for (int off = 16; off; off >>= 1) s += __shfl_xor(s, off, 32);
      if (lane == 0) scores[row] = s;
    }
    __syncthreads();

    // (c1) softmax over L, per batch column
    if (tid < 16) {
      int b = tid;
      float mx = -1e30f;
      for (int l = 0; l < L; ++l) mx = fmaxf(mx, scores[l * 16 + b]);
      float sum = 0.0f;
      for (int l = 0; l < L; ++l) {
        float e = expf(scores[l * 16 + b] - mx);
        alphL[l * 16 + b] = e;
        sum += e;
      }
      float inv = 1.0f / sum;
      for (int l = 0; l < L; ++l) {
        float v = alphL[l * 16 + b] * inv;
        alphL[l * 16 + b] = v;
        a.alpha[g][(size_t)t * L * 16 + l * 16 + b] = v;
      }
    }
    __syncthreads();

    // (c2) ct[b,:] = sum_l alpha[l,b] * inH[l,b,:]
    for (int i = tid; i < 16 * 1024; i += 256) {
      int b = i >> 10, hh = i & 1023;
      float s = 0.0f;
      for (int l = 0; l < L; ++l)
        s += alphL[l * 16 + b] * inH[((size_t)l * 16 + b) * 1024 + hh];
      a.cOut[g][((size_t)t * 16 + b) * 1024 + hh] = s;
    }
    __syncthreads();

    // (d) gates = t_emb[t] @ dWih^T + h @ dWhh^T
    const float* xt = a.temb + (size_t)t * 16 * 256;
#pragma unroll 1
    for (int nt = 0; nt < 16; ++nt) {
      int n0 = wid * 256 + nt * 16;
      v8f acc = {};
#pragma unroll
      for (int k0 = 0; k0 < 256; k0 += 32)
        acc = wmma_bf16(load_a_f32(xt, 256, 0, k0), load_b_w(a.dWih, 256, n0, k0), acc);
#pragma unroll
      for (int k0 = 0; k0 < 512; k0 += 32)
        acc = wmma_bf16(load_a_lds(hS, k0), load_b_w(a.dWhh, 512, n0, k0), acc);
      int hf = lane >> 4, n = lane & 15;
#pragma unroll
      for (int r = 0; r < 8; ++r)
        gates[(hf * 8 + r) * 2048 + n0 + n] = acc[r];
    }
    __syncthreads();

    // (e) LSTMCell update
#pragma unroll 1
    for (int u = 0; u < 32; ++u) {
      int idx = tid * 32 + u;
      int b = idx >> 9, j = idx & 511;
      const float* gb = gates + b * 2048;
      float gi = gb[j]        + a.db[j];
      float gf = gb[512 + j]  + a.db[512 + j];
      float gg = gb[1024 + j] + a.db[1024 + j];
      float go = gb[1536 + j] + a.db[1536 + j];
      float c = sigf(gf) * creg[u] + sigf(gi) * tanhf(gg);
      creg[u] = c;
      float h = sigf(go) * tanhf(c);
      a.sOut[g][((size_t)t * 16 + b) * 512 + j] = h;
      hS[b * 512 + j] = (__bf16)h;
    }
    __syncthreads();
  }
}

// ---------------------------------------------------------------------------
// gamma/lam scalar gates: one wave per (t,b) row.
// ---------------------------------------------------------------------------
__global__ __launch_bounds__(256) void gate_scalars(
    const float* __restrict__ ck, const float* __restrict__ cx,
    const float* __restrict__ sx, const float* __restrict__ temb,
    const float* __restrict__ w_d, const float* __restrict__ u_d,
    const float* __restrict__ v_d, const float* __restrict__ w_e,
    const float* __restrict__ u_e, const float* __restrict__ v_e,
    float* __restrict__ gam, float* __restrict__ lam) {
  int row = blockIdx.x * 8 + (threadIdx.x >> 5);
  if (row >= 512) return;
  int lane = threadIdx.x & 31;
  float ag = 0.0f, al = 0.0f;
  for (int i = lane; i < 1024; i += 32) {
    float cxi = cx[(size_t)row * 1024 + i];
    ag += (ck[(size_t)row * 1024 + i] + cxi) * w_d[i];
    al += cxi * w_e[i];
  }
  for (int i = lane; i < 512; i += 32) {
    float s = sx[(size_t)row * 512 + i];
    ag += s * u_d[i];
    al += s * u_e[i];
  }
  for (int i = lane; i < 256; i += 32) {
    float e = temb[(size_t)row * 256 + i];
    ag += e * v_d[i];
    al += e * v_e[i];
  }
#pragma unroll
  for (int off = 16; off; off >>= 1) {
    ag += __shfl_xor(ag, off, 32);
    al += __shfl_xor(al, off, 32);
  }
  if (lane == 0) { gam[row] = sigf(ag); lam[row] = sigf(al); }
}

// ---------------------------------------------------------------------------
// In-place masked softmax over vocab, scaled by lam: prob = lam * softmax.
// Vocab ids >= 16000 are masked (-inf in reference) -> 0.
// ---------------------------------------------------------------------------
__global__ __launch_bounds__(256) void softmax_scale(float* __restrict__ prob,
                                                     const float* __restrict__ lam) {
  __shared__ float red[8];
  int row = blockIdx.x;
  int tid = threadIdx.x, wid = tid >> 5, lane = tid & 31;
  float* p = prob + (size_t)row * 32000;
  const int CV = 16000;

  float mx = -1e30f;
  for (int i = tid; i < CV; i += 256) mx = fmaxf(mx, p[i]);
#pragma unroll
  for (int off = 16; off; off >>= 1) mx = fmaxf(mx, __shfl_xor(mx, off, 32));
  if (lane == 0) red[wid] = mx;
  __syncthreads();
  mx = red[0];
#pragma unroll
  for (int w = 1; w < 8; ++w) mx = fmaxf(mx, red[w]);
  __syncthreads();

  float sum = 0.0f;
  for (int i = tid; i < CV; i += 256) sum += expf(p[i] - mx);
#pragma unroll
  for (int off = 16; off; off >>= 1) sum += __shfl_xor(sum, off, 32);
  if (lane == 0) red[wid] = sum;
  __syncthreads();
  sum = 0.0f;
#pragma unroll
  for (int w = 0; w < 8; ++w) sum += red[w];

  float scale = lam[row] / sum;
  for (int i = tid; i < CV; i += 256) p[i] = expf(p[i] - mx) * scale;
  for (int i = CV + tid; i < 32000; i += 256) p[i] = 0.0f;
}

// ---------------------------------------------------------------------------
// Pointer-copy scatter adds.
// ---------------------------------------------------------------------------
__global__ void scatter_src(float* __restrict__ prob, const float* __restrict__ ax,
                            const int* __restrict__ sidx,
                            const float* __restrict__ gam, const float* __restrict__ lam) {
  int i = blockIdx.x * 256 + threadIdx.x;  // T*S*B = 32*96*16
  if (i >= 32 * 96 * 16) return;
  int b = i & 15;
  int l = (i >> 4) % 96;
  int t = i / (96 * 16);
  int tb = t * 16 + b;
  float w = (1.0f - lam[tb]) * gam[tb] * ax[i];
  atomicAdd(&prob[(size_t)tb * 32000 + sidx[l * 16 + b]], w);
}

__global__ void scatter_val(float* __restrict__ prob, const float* __restrict__ ak,
                            const float* __restrict__ av, const int* __restrict__ vidx,
                            const float* __restrict__ gam, const float* __restrict__ lam) {
  int i = blockIdx.x * 256 + threadIdx.x;  // T*VL*B = 32*96*16
  if (i >= 32 * 96 * 16) return;
  int b = i & 15;
  int l = (i >> 4) % 96;
  int t = i / (96 * 16);
  int tb = t * 16 + b;
  int attr = l / 6;  // LV = 6
  float w = (1.0f - lam[tb]) * (1.0f - gam[tb]) *
            ak[(size_t)(t * 16 + attr) * 16 + b] * av[i];
  atomicAdd(&prob[(size_t)tb * 32000 + vidx[l * 16 + b]], w);
}

__global__ void loss_k(const float* __restrict__ prob, const int* __restrict__ tidx,
                       float* __restrict__ loss) {
  int b = threadIdx.x;
  if (b >= 16) return;
  float s = 0.0f;
  for (int t = 0; t < 32; ++t)
    s += -logf(prob[(size_t)(t * 16 + b) * 32000 + tidx[t * 16 + b]]);
  loss[b] = s / 32.0f;
}

// ---------------------------------------------------------------------------
// Host orchestration
// ---------------------------------------------------------------------------
extern "C" void kernel_launch(void* const* d_in, const int* in_sizes, int n_in,
                              void* d_out, int out_size, void* d_ws, size_t ws_size,
                              hipStream_t stream) {
  (void)in_sizes; (void)n_in; (void)out_size; (void)ws_size;
  constexpr int E = 256, H = 512, C = 32000, B = 16, S = 96, T = 32;

  const float* k_emb = (const float*)d_in[0];
  const float* v_emb = (const float*)d_in[1];
  const float* s_emb = (const float*)d_in[2];
  const float* t_emb = (const float*)d_in[3];
  const int P = 4;  // params start (dict insertion order)
  const int* s_idx = (const int*)d_in[43];
  const int* v_idx = (const int*)d_in[44];
  const int* t_idx = (const int*)d_in[45];

  // workspace bump allocator
  char* base = (char*)d_ws;
  size_t off = 0;
  auto alloc = [&](size_t bytes) -> void* {
    void* p = base + off;
    off = (off + bytes + 255) & ~(size_t)255;
    return p;
  };
  auto mkbf = [&](const float* src, size_t n) -> __bf16* {
    __bf16* d = (__bf16*)alloc(n * sizeof(__bf16));
    cvt_f32_bf16_k<<<(unsigned)((n + 255) / 256), 256, 0, stream>>>(src, d, (int)n);
    return d;
  };
  auto fbuf = [&](size_t n) -> float* { return (float*)alloc(n * sizeof(float)); };

  // --- bf16 weights ---
  const __bf16 *encWih[6], *encWhh[6];
  const float*  encB[6];
  for (int e = 0; e < 3; ++e)
    for (int d = 0; d < 2; ++d) {
      int pi = P + e * 6 + d * 3;
      encWih[e * 2 + d] = mkbf((const float*)d_in[pi + 0], (size_t)4 * H * E);
      encWhh[e * 2 + d] = mkbf((const float*)d_in[pi + 1], (size_t)4 * H * H);
      encB[e * 2 + d]   = (const float*)d_in[pi + 2];
    }
  const __bf16* dWih_bf = mkbf((const float*)d_in[P + 18], (size_t)4 * H * E);
  const __bf16* dWhh_bf = mkbf((const float*)d_in[P + 19], (size_t)4 * H * H);
  const float*  dec_b   = (const float*)d_in[P + 20];
  const __bf16* sInitW = mkbf((const float*)d_in[P + 21], (size_t)H * 2 * H);
  const float*  sInitB = (const float*)d_in[P + 22];
  const __bf16* kInitW = mkbf((const float*)d_in[P + 23], (size_t)H * 2 * H);
  const float*  kInitB = (const float*)d_in[P + 24];
  const __bf16* vInitW = mkbf((const float*)d_in[P + 25], (size_t)H * 2 * H);
  const float*  vInitB = (const float*)d_in[P + 26];
  const __bf16* wA_bf = mkbf((const float*)d_in[P + 27], (size_t)H * 2 * H);
  const __bf16* vA_bf = mkbf((const float*)d_in[P + 28], (size_t)H * H);
  const float*  u_a   = (const float*)d_in[P + 29];
  const __bf16* wB_bf = mkbf((const float*)d_in[P + 30], (size_t)H * H);
  const __bf16* vB_bf = mkbf((const float*)d_in[P + 31], (size_t)H * 2 * H);
  const __bf16* uB_bf = mkbf((const float*)d_in[P + 32], (size_t)C * H);
  const float* w_d = (const float*)d_in[P + 33];
  const float* u_d = (const float*)d_in[P + 34];
  const float* v_d = (const float*)d_in[P + 35];
  const float* w_e = (const float*)d_in[P + 36];
  const float* u_e = (const float*)d_in[P + 37];
  const float* v_e = (const float*)d_in[P + 38];

  // --- f32 buffers ---
  float* hcat_k = fbuf((size_t)64 * 16 * 1024);
  float* hcat_v = fbuf((size_t)96 * 16 * 1024);
  float* hcat_s = fbuf((size_t)96 * 16 * 1024);
  float* cell_k = fbuf(16 * 1024);
  float* cell_v = fbuf(16 * 1024);
  float* cell_s = fbuf(16 * 1024);
  float* c0_s = fbuf(16 * 512); float* h0_s = fbuf(16 * 512);
  float* c0_k = fbuf(16 * 512); float* h0_k = fbuf(16 * 512);
  float* c0_v = fbuf(16 * 512); float* h0_v = fbuf(16 * 512);
  float* hkattr = fbuf((size_t)16 * 16 * 1024);
  float* wa_s = fbuf((size_t)96 * 16 * 512);
  float* wa_k = fbuf((size_t)16 * 16 * 512);
  float* wa_v = fbuf((size_t)96 * 16 * 512);
  float* al_x = fbuf((size_t)32 * 96 * 16);
  float* al_k = fbuf((size_t)32 * 16 * 16);
  float* al_v = fbuf((size_t)32 * 96 * 16);
  float* cx = fbuf((size_t)32 * 16 * 1024);
  float* ck = fbuf((size_t)32 * 16 * 1024);
  float* cv = fbuf((size_t)32 * 16 * 1024);
  float* sx = fbuf((size_t)32 * 16 * 512);
  float* sk = fbuf((size_t)32 * 16 * 512);
  float* sv = fbuf((size_t)32 * 16 * 512);
  float* pre = fbuf((size_t)512 * 512);
  float* gam = fbuf(512);
  float* lam = fbuf(512);

  float* loss = (float*)d_out;
  float* prob = (float*)d_out + 16;

  auto gemm = [&](const float* A, const __bf16* W, const float* bias, float* Cp,
                  int M, int N, int K, int accum) {
    dim3 gr((unsigned)((M + 15) / 16), (unsigned)((N + 127) / 128));
    gemm_aw<<<gr, 256, 0, stream>>>(A, W, bias, Cp, M, N, K, accum);
  };

  // --- 1) Encoders (6 persistent workgroups) ---
  const int ENC_LDS = 16 * 512 * 2 + 16 * 2048 * 4;  // 147456
  EncArgs ea{};
  const int   encL[3] = {64, 96, 96};
  const float* encX[3] = {k_emb, v_emb, s_emb};
  float* hcats[3] = {hcat_k, hcat_v, hcat_s};
  float* cells[3] = {cell_k, cell_v, cell_s};
  for (int e = 0; e < 3; ++e)
    for (int d = 0; d < 2; ++d) {
      int g = e * 2 + d;
      ea.x[g] = encX[e];
      ea.Wih[g] = encWih[g];
      ea.Whh[g] = encWhh[g];
      ea.bias[g] = encB[g];
      ea.hcat[g] = hcats[e];
      ea.cell[g] = cells[e];
      ea.L[g] = encL[e];
      ea.dir[g] = d;
    }
  (void)hipFuncSetAttribute((const void*)lstm_enc,
                            hipFuncAttributeMaxDynamicSharedMemorySize, ENC_LDS);
  lstm_enc<<<6, 256, ENC_LDS, stream>>>(ea);

  // --- 2) Decoder initial states: c0 = cell @ init_w^T + b; h0 = tanh(c0) ---
  gemm(cell_s, sInitW, sInitB, c0_s, 16, 512, 1024, 0);
  gemm(cell_k, kInitW, kInitB, c0_k, 16, 512, 1024, 0);
  gemm(cell_v, vInitW, vInitB, c0_v, 16, 512, 1024, 0);
  tanh_k<<<(16 * 512 + 255) / 256, 256, 0, stream>>>(c0_s, h0_s, 16 * 512);
  tanh_k<<<(16 * 512 + 255) / 256, 256, 0, stream>>>(c0_k, h0_k, 16 * 512);
  tanh_k<<<(16 * 512 + 255) / 256, 256, 0, stream>>>(c0_v, h0_v, 16 * 512);

  // --- 3) h_k_attr + attention pre-projections wa_h = input_h @ w_a^T ---
  hk_attr_k<<<(16 * 16 * 1024) / 256, 256, 0, stream>>>(hcat_k, hkattr);
  gemm(hcat_s, wA_bf, nullptr, wa_s, 96 * 16, 512, 1024, 0);
  gemm(hkattr, wA_bf, nullptr, wa_k, 16 * 16, 512, 1024, 0);
  gemm(hcat_v, wA_bf, nullptr, wa_v, 96 * 16, 512, 1024, 0);

  // --- 4) Decoders (3 persistent workgroups) ---
  const int DEC_LDS = 16 * 512 * 2 + 16 * 2048 * 4 + 2 * 96 * 16 * 4;  // 159744
  DecArgs da{};
  da.h0[0] = h0_s; da.c0[0] = c0_s; da.waH[0] = wa_s; da.inH[0] = hcat_s;
  da.alpha[0] = al_x; da.cOut[0] = cx; da.sOut[0] = sx; da.L[0] = 96;
  da.h0[1] = h0_k; da.c0[1] = c0_k; da.waH[1] = wa_k; da.inH[1] = hkattr;
  da.alpha[1] = al_k; da.cOut[1] = ck; da.sOut[1] = sk; da.L[1] = 16;
  da.h0[2] = h0_v; da.c0[2] = c0_v; da.waH[2] = wa_v; da.inH[2] = hcat_v;
  da.alpha[2] = al_v; da.cOut[2] = cv; da.sOut[2] = sv; da.L[2] = 96;
  da.v_a = vA_bf; da.u_a = u_a;
  da.dWih = dWih_bf; da.dWhh = dWhh_bf; da.db = dec_b; da.temb = t_emb;
  (void)hipFuncSetAttribute((const void*)lstm_dec,
                            hipFuncAttributeMaxDynamicSharedMemorySize, DEC_LDS);
  lstm_dec<<<3, 256, DEC_LDS, stream>>>(da);

  // --- 5) Generation logits: tanh(sx@w_b^T + cx@v_b^T) @ u_b^T ---
  gemm(sx, wB_bf, nullptr, pre, 512, 512, 512, 0);
  gemm(cx, vB_bf, nullptr, pre, 512, 512, 1024, 1);
  tanh_k<<<(512 * 512) / 256, 256, 0, stream>>>(pre, pre, 512 * 512);
  gemm(pre, uB_bf, nullptr, prob, 512, 32000, 512, 0);  // logits into d_out

  // --- 6) gamma/lam, masked softmax (scaled by lam), pointer scatters, loss ---
  gate_scalars<<<64, 256, 0, stream>>>(ck, cx, sx, t_emb, w_d, u_d, v_d,
                                       w_e, u_e, v_e, gam, lam);
  softmax_scale<<<512, 256, 0, stream>>>(prob, lam);
  scatter_src<<<(32 * 96 * 16) / 256, 256, 0, stream>>>(prob, al_x, s_idx, gam, lam);
  scatter_val<<<(32 * 96 * 16) / 256, 256, 0, stream>>>(prob, al_k, al_v, v_idx, gam, lam);
  loss_k<<<1, 32, 0, stream>>>(prob, t_idx, loss);
}
